// JointSentenceBiLSTM2_WithPostag_81114752352626
// MI455X (gfx1250) — matrix operand order repaired and checked by reading, not compiled, not run on hardware
//
#include <hip/hip_runtime.h>
#include <hip/hip_bf16.h>
#include <math.h>

typedef __attribute__((ext_vector_type(16))) __bf16 v16bf;
typedef __attribute__((ext_vector_type(8)))  __bf16 v8bf;
typedef __attribute__((ext_vector_type(8)))  float  v8f;

#define B_    64
#define L_    128
#define E_    34
#define A_    36
#define WD_   300
#define PD_   50
#define H_    256
#define D_    350
#define DPAD  352        // K-pad of 350 to multiple of 32 (352 = 11*32)
#define HS_   512        // 2H
#define NPAD  48         // N-pad for E=34 / A=36 heads (3 WMMA tiles)

// ---------- helpers ----------
__device__ __forceinline__ unsigned short f2bf(float x) {
    union { float f; unsigned u; } cv; cv.f = x;
    unsigned u = cv.u;
    u += 0x7FFFu + ((u >> 16) & 1u);          // round-to-nearest-even
    return (unsigned short)(u >> 16);
}
__device__ __forceinline__ float sigm(float x) { return 1.0f / (1.0f + __expf(-x)); }

#define WMMA_BF16(a, b, c) \
    __builtin_amdgcn_wmma_f32_16x16x32_bf16(false, (a), false, (b), (short)0, (c), false, false)

// ============================================================================
// Narrow GEMM: one wave -> one 16x16 tile.  Used for the latency-bound
// sequential recurrent steps (maximize wave parallelism) and N%64!=0 heads.
// C[M,N] = A[M,K] * W[N,K]^T (+Cadd)(+bias0+bias1); bf16 in, f32 out.
// A/B fragments use contiguous k order (lane half h holds k 16h..16h+15).
// ============================================================================
__global__ __launch_bounds__(128)
void wmma_gemm_n16(const unsigned short* __restrict__ A, int lda,
                   const unsigned short* __restrict__ W, int ldw,
                   const float* __restrict__ Cadd, long ldadd,
                   const float* __restrict__ bias0,
                   const float* __restrict__ bias1,
                   float* __restrict__ C, int ldc,
                   int M, int N, int K)
{
    const int wave = threadIdx.x >> 5;
    const int lane = threadIdx.x & 31;
    const int tile = blockIdx.x * (blockDim.x >> 5) + wave;
    const int ntn  = N >> 4;
    const int nTiles = (M >> 4) * ntn;
    if (tile >= nTiles) return;                 // wave-uniform exit
    const int tm = (tile / ntn) << 4;
    const int tn = (tile % ntn) << 4;

    const int lo  = (lane < 16) ? 1 : 0;
    const int mA  = tm + (lane & 15);           // A row for this lane
    const int nB  = tn + (lane & 15);           // B column (= W row) for this lane
    const int kbA = lo ? 0 : 16;                // A: lanes 0-15 hold K 0..15
    const int kbB = lo ? 0 : 16;                // B: lanes 0-15 hold K 0..15

    v8f acc = {};
    union { v16bf v; v8bf h[2]; } af;
    for (int k0 = 0; k0 < K; k0 += 32) {
        const unsigned short* pa = A + (long)mA * lda + k0 + kbA;
        af.h[0] = *reinterpret_cast<const v8bf*>(pa);
        af.h[1] = *reinterpret_cast<const v8bf*>(pa + 8);
        const unsigned short* pb = W + (long)nB * ldw + k0 + kbB;
        v16bf bf = *reinterpret_cast<const v16bf*>(pb);
        acc = WMMA_BF16(af.v, bf, acc);
    }

    float bias = 0.0f;
    if (bias0) bias += bias0[nB];
    if (bias1) bias += bias1[nB];
    const int mBase = tm + (lo ? 0 : 8);        // C: VGPR j -> row j / j+8
    for (int j = 0; j < 8; ++j) {
        const int m = mBase + j;
        float v = acc[j] + bias;
        if (Cadd) v += Cadd[(long)m * ldadd + nB];
        C[(long)m * ldc + nB] = v;
    }
}

// ============================================================================
// Wide GEMM: one wave -> 32x64 tile (2 A-frags x 4 B-frags -> 8 WMMAs / K-step).
// ~2.7x fewer bytes moved per WMMA than the 16x16 version; used for the
// large bandwidth-bound GEMMs (M%32==0, N%64==0).
// ============================================================================
__global__ __launch_bounds__(128)
void wmma_gemm_m32n64(const unsigned short* __restrict__ A, int lda,
                      const unsigned short* __restrict__ W, int ldw,
                      const float* __restrict__ bias0,
                      const float* __restrict__ bias1,
                      float* __restrict__ C, int ldc,
                      int M, int N, int K)
{
    const int wave = threadIdx.x >> 5;
    const int lane = threadIdx.x & 31;
    const int tile = blockIdx.x * (blockDim.x >> 5) + wave;
    const int ntn  = N >> 6;                    // 64-wide N groups
    const int nTiles = (M >> 5) * ntn;
    if (tile >= nTiles) return;
    const int tm = (tile / ntn) << 5;
    const int tn = (tile % ntn) << 6;

    const int lo  = (lane < 16) ? 1 : 0;
    const int mA  = tm + (lane & 15);
    const int nB  = tn + (lane & 15);
    const int kbA = lo ? 0 : 16;
    const int kbB = lo ? 0 : 16;

    v8f acc00 = {}, acc01 = {}, acc02 = {}, acc03 = {};
    v8f acc10 = {}, acc11 = {}, acc12 = {}, acc13 = {};
    union { v16bf v; v8bf h[2]; } a0, a1;
    for (int k0 = 0; k0 < K; k0 += 32) {
        const unsigned short* pa = A + (long)mA * lda + k0 + kbA;
        a0.h[0] = *reinterpret_cast<const v8bf*>(pa);
        a0.h[1] = *reinterpret_cast<const v8bf*>(pa + 8);
        const unsigned short* pa1 = pa + (long)16 * lda;
        a1.h[0] = *reinterpret_cast<const v8bf*>(pa1);
        a1.h[1] = *reinterpret_cast<const v8bf*>(pa1 + 8);
        const unsigned short* pb = W + (long)nB * ldw + k0 + kbB;
        v16bf b0 = *reinterpret_cast<const v16bf*>(pb);
        v16bf b1 = *reinterpret_cast<const v16bf*>(pb + (long)16 * ldw);
        v16bf b2 = *reinterpret_cast<const v16bf*>(pb + (long)32 * ldw);
        v16bf b3 = *reinterpret_cast<const v16bf*>(pb + (long)48 * ldw);
        acc00 = WMMA_BF16(a0.v, b0, acc00);
        acc01 = WMMA_BF16(a0.v, b1, acc01);
        acc02 = WMMA_BF16(a0.v, b2, acc02);
        acc03 = WMMA_BF16(a0.v, b3, acc03);
        acc10 = WMMA_BF16(a1.v, b0, acc10);
        acc11 = WMMA_BF16(a1.v, b1, acc11);
        acc12 = WMMA_BF16(a1.v, b2, acc12);
        acc13 = WMMA_BF16(a1.v, b3, acc13);
    }

    v8f* accs[2][4] = { { &acc00, &acc01, &acc02, &acc03 },
                        { &acc10, &acc11, &acc12, &acc13 } };
    for (int mi = 0; mi < 2; ++mi) {
        const int mBase = tm + mi * 16 + (lo ? 0 : 8);
        for (int s = 0; s < 4; ++s) {
            const int n = nB + 16 * s;
            float bias = 0.0f;
            if (bias0) bias += bias0[n];
            if (bias1) bias += bias1[n];
            v8f& acc = *accs[mi][s];
            for (int j = 0; j < 8; ++j)
                C[(long)(mBase + j) * ldc + n] = acc[j] + bias;
        }
    }
}

// ---------- embedding gather -> bf16 K-padded rows [B*L, DPAD] ----------
__global__ void embed_pack(const int* __restrict__ ids, const int* __restrict__ pids,
                           const float* __restrict__ wtab, const float* __restrict__ ptab,
                           unsigned short* __restrict__ out)
{
    int idx = blockIdx.x * blockDim.x + threadIdx.x;
    const int total = B_ * L_ * DPAD;
    if (idx >= total) return;
    int row = idx / DPAD, c = idx - row * DPAD;
    float v = 0.0f;
    if (c < WD_)      v = wtab[(long)ids[row]  * WD_ + c];
    else if (c < D_)  v = ptab[(long)pids[row] * PD_ + (c - WD_)];
    out[idx] = f2bf(v);
}

// ---------- fp32 weight -> zero-padded bf16 [Npad, Kpad] ----------
__global__ void pack_w(const float* __restrict__ src, int srcLd, int colOff,
                       int N, int K, unsigned short* __restrict__ dst,
                       int Kpad, int Npad)
{
    int idx = blockIdx.x * blockDim.x + threadIdx.x;
    const int total = Npad * Kpad;
    if (idx >= total) return;
    int n = idx / Kpad, k = idx - n * Kpad;
    float v = (n < N && k < K) ? src[(long)n * srcLd + colOff + k] : 0.0f;
    dst[idx] = f2bf(v);
}

__global__ void zero_f32(float* p, int n) {
    int i = blockIdx.x * blockDim.x + threadIdx.x;
    if (i < n) p[i] = 0.0f;
}

// ---------- LSTM pointwise update (both directions), t = step index ----------
__global__ void lstm_update(const float* __restrict__ gates,   // [2][B][4H]
                            float* __restrict__ c_ws,          // [2][B][H]
                            unsigned short* __restrict__ h_bf, // [2][B][H] bf16
                            unsigned short* __restrict__ hs_bf,// [B*L][2H] bf16
                            int t)
{
    int idx = blockIdx.x * blockDim.x + threadIdx.x;
    if (idx >= 2 * B_ * H_) return;
    int dir = idx / (B_ * H_);
    int r   = idx - dir * (B_ * H_);
    int b = r / H_, j = r - b * H_;
    const float* g = gates + (long)dir * B_ * 4 * H_ + (long)b * 4 * H_;
    float gi = g[j], gf = g[H_ + j], gg = g[2 * H_ + j], go = g[3 * H_ + j];
    float c  = c_ws[idx];
    float cn = sigm(gf) * c + sigm(gi) * tanhf(gg);
    float h  = sigm(go) * tanhf(cn);
    c_ws[idx] = cn;
    unsigned short hb = f2bf(h);
    h_bf[idx] = hb;
    int te = (dir == 0) ? t : (L_ - 1 - t);
    hs_bf[((long)(b * L_ + te)) * HS_ + dir * H_ + j] = hb;
}

// ---------- decoder event head, one step i (sequential dependency on g_trg) ----------
__global__ void ev_step(const float* __restrict__ ev_base,  // [B*L][NPAD]
                        const float* __restrict__ We,       // [E][545] fp32
                        const float* __restrict__ be,
                        float* __restrict__ g_trg,          // [B][E-1]
                        int* __restrict__ ev_active,        // [B]
                        float* __restrict__ out_ev, int i)
{
    int b = threadIdx.x;
    if (b >= B_) return;
    float gt[E_ - 1];
    for (int j = 0; j < E_ - 1; ++j) gt[j] = g_trg[b * (E_ - 1) + j];
    float best = -3.4e38f; int pred = 0;
    for (int e = 0; e < E_; ++e) {
        float v = ev_base[(long)(b * L_ + i) * NPAD + e] + be[e];
        const float* wrow = We + (long)e * (2 * H_ + E_ - 1) + 2 * H_;
        for (int j = 0; j < E_ - 1; ++j) v += gt[j] * wrow[j];
        out_ev[(long)(b * L_ + i) * E_ + e] = v;
        if (v > best) { best = v; pred = e; }   // strict > keeps first-max (argmax tie rule)
    }
    int act = (pred > 0) ? 1 : 0;
    if (act) g_trg[b * (E_ - 1) + (pred - 1)] = 1.0f;
    ev_active[b] = act;
}

// ---------- decoder argument head, one step i ----------
__global__ __launch_bounds__(256)
void arg_step(const float* __restrict__ base_arg,  // [B*L][NPAD]  hs@Wa1^T
              const float* __restrict__ trig_proj, // [B*L][NPAD]  hs@Wa2^T
              const float* __restrict__ Wa,        // [A][1059] fp32
              const float* __restrict__ ba,
              float* __restrict__ g_arg,           // [B*L][A-1]
              const int* __restrict__ ev_active,
              float* __restrict__ out_ar, int i)
{
    __shared__ float sW[A_ * (A_ - 1)];
    __shared__ float sB[A_];
    for (int idx = threadIdx.x; idx < A_ * (A_ - 1); idx += blockDim.x) {
        int a = idx / (A_ - 1), j = idx - a * (A_ - 1);
        sW[idx] = Wa[(long)a * (4 * H_ + A_ - 1) + 4 * H_ + j];
    }
    for (int idx = threadIdx.x; idx < A_; idx += blockDim.x) sB[idx] = ba[idx];
    __syncthreads();

    int row = blockIdx.x * blockDim.x + threadIdx.x;
    if (row >= B_ * L_) return;
    int b = row >> 7, l = row & (L_ - 1);
    float ga[A_ - 1];
    for (int j = 0; j < A_ - 1; ++j) ga[j] = g_arg[(long)row * (A_ - 1) + j];
    const float* bp = base_arg  + (long)row * NPAD;
    const float* tp = trig_proj + (long)(b * L_ + i) * NPAD;
    float* outp = out_ar + ((long)(b * L_ + i) * L_ + l) * A_;
    float best = -3.4e38f; int pred = 0;
    for (int a = 0; a < A_; ++a) {
        float v = bp[a] + tp[a] + sB[a];
        const float* w = sW + a * (A_ - 1);
        for (int j = 0; j < A_ - 1; ++j) v += ga[j] * w[j];
        outp[a] = v;
        if (v > best) { best = v; pred = a; }
    }
    if (ev_active[b] && pred > 0) g_arg[(long)row * (A_ - 1) + pred - 1] = 1.0f;
}

// ---------- host launcher ----------
static inline void launch_gemm(const unsigned short* A, int lda,
                               const unsigned short* W, int ldw,
                               const float* Cadd, long ldadd,
                               const float* b0, const float* b1,
                               float* C, int ldc, int M, int N, int K,
                               bool wide, hipStream_t s)
{
    if (wide && (M % 32 == 0) && (N % 64 == 0) && Cadd == nullptr) {
        int tiles  = (M / 32) * (N / 64);
        int blocks = (tiles + 3) / 4;
        wmma_gemm_m32n64<<<blocks, 128, 0, s>>>(A, lda, W, ldw, b0, b1, C, ldc, M, N, K);
    } else {
        int tiles  = (M / 16) * (N / 16);
        int blocks = (tiles + 3) / 4;
        wmma_gemm_n16<<<blocks, 128, 0, s>>>(A, lda, W, ldw, Cadd, ldadd, b0, b1, C, ldc, M, N, K);
    }
}

extern "C" void kernel_launch(void* const* d_in, const int* in_sizes, int n_in,
                              void* d_out, int out_size, void* d_ws, size_t ws_size,
                              hipStream_t stream)
{
    (void)in_sizes; (void)n_in; (void)out_size; (void)ws_size;
    const int*   input_ids  = (const int*)d_in[0];
    const int*   postag_ids = (const int*)d_in[1];
    const float* wtab  = (const float*)d_in[2];
    const float* ptab  = (const float*)d_in[3];
    const float* Wih_f = (const float*)d_in[4];
    const float* Whh_f = (const float*)d_in[5];
    const float* bih_f = (const float*)d_in[6];
    const float* bhh_f = (const float*)d_in[7];
    const float* Wih_b = (const float*)d_in[8];
    const float* Whh_b = (const float*)d_in[9];
    const float* bih_b = (const float*)d_in[10];
    const float* bhh_b = (const float*)d_in[11];
    const float* We    = (const float*)d_in[12];
    const float* be    = (const float*)d_in[13];
    const float* Wa    = (const float*)d_in[14];
    const float* ba    = (const float*)d_in[15];

    float* out_ev = (float*)d_out;                         // [B,L,E]
    float* out_ar = out_ev + (long)B_ * L_ * E_;           // [B,L,L,A]

    // workspace carve-up (256B aligned)
    char* w = (char*)d_ws;
    auto alloc = [&](size_t bytes) {
        char* p = w; w += (bytes + 255) & ~(size_t)255; return p;
    };
    const int ROWS = B_ * L_;                              // 8192
    unsigned short* emb    = (unsigned short*)alloc((size_t)ROWS * DPAD * 2);
    unsigned short* wWihF  = (unsigned short*)alloc((size_t)4 * H_ * DPAD * 2);
    unsigned short* wWihB  = (unsigned short*)alloc((size_t)4 * H_ * DPAD * 2);
    unsigned short* wWhhF  = (unsigned short*)alloc((size_t)4 * H_ * H_ * 2);
    unsigned short* wWhhB  = (unsigned short*)alloc((size_t)4 * H_ * H_ * 2);
    unsigned short* wWe1   = (unsigned short*)alloc((size_t)NPAD * HS_ * 2);
    unsigned short* wWa1   = (unsigned short*)alloc((size_t)NPAD * HS_ * 2);
    unsigned short* wWa2   = (unsigned short*)alloc((size_t)NPAD * HS_ * 2);
    float*          xgF    = (float*)alloc((size_t)ROWS * 4 * H_ * 4);
    float*          xgB    = (float*)alloc((size_t)ROWS * 4 * H_ * 4);
    float*          gates  = (float*)alloc((size_t)2 * B_ * 4 * H_ * 4);
    unsigned short* h_bf   = (unsigned short*)alloc((size_t)2 * B_ * H_ * 2);
    float*          c_ws   = (float*)alloc((size_t)2 * B_ * H_ * 4);
    unsigned short* hs_bf  = (unsigned short*)alloc((size_t)ROWS * HS_ * 2);
    float*          evbase = (float*)alloc((size_t)ROWS * NPAD * 4);
    float*          bsarg  = (float*)alloc((size_t)ROWS * NPAD * 4);
    float*          tproj  = (float*)alloc((size_t)ROWS * NPAD * 4);
    float*          g_trg  = (float*)alloc((size_t)B_ * (E_ - 1) * 4);
    float*          g_arg  = (float*)alloc((size_t)ROWS * (A_ - 1) * 4);
    int*            evact  = (int*)alloc((size_t)B_ * 4);

    // 1) embedding gather + weight conversion to padded bf16
    {
        int tot = ROWS * DPAD;
        embed_pack<<<(tot + 255) / 256, 256, 0, stream>>>(input_ids, postag_ids, wtab, ptab, emb);
    }
    auto packW = [&](const float* src, int srcLd, int colOff, int N, int K,
                     unsigned short* dst, int Kpad, int Npad) {
        int tot = Npad * Kpad;
        pack_w<<<(tot + 255) / 256, 256, 0, stream>>>(src, srcLd, colOff, N, K, dst, Kpad, Npad);
    };
    packW(Wih_f, D_, 0, 4 * H_, D_, wWihF, DPAD, 4 * H_);
    packW(Wih_b, D_, 0, 4 * H_, D_, wWihB, DPAD, 4 * H_);
    packW(Whh_f, H_, 0, 4 * H_, H_, wWhhF, H_, 4 * H_);
    packW(Whh_b, H_, 0, 4 * H_, H_, wWhhB, H_, 4 * H_);
    packW(We, 2 * H_ + E_ - 1, 0,   E_, HS_, wWe1, HS_, NPAD);
    packW(Wa, 4 * H_ + A_ - 1, 0,   A_, HS_, wWa1, HS_, NPAD);
    packW(Wa, 4 * H_ + A_ - 1, HS_, A_, HS_, wWa2, HS_, NPAD);

    // 2) zero-init states
    zero_f32<<<(2 * B_ * H_ + 255) / 256, 256, 0, stream>>>(c_ws, 2 * B_ * H_);
    zero_f32<<<((2 * B_ * H_ / 2) + 255) / 256, 256, 0, stream>>>((float*)h_bf, 2 * B_ * H_ / 2);
    zero_f32<<<(B_ * (E_ - 1) + 255) / 256, 256, 0, stream>>>(g_trg, B_ * (E_ - 1));
    zero_f32<<<(ROWS * (A_ - 1) + 255) / 256, 256, 0, stream>>>(g_arg, ROWS * (A_ - 1));

    // 3) input projection xg = emb @ Wih^T + bih + bhh  (wide tile: bandwidth-bound)
    launch_gemm(emb, DPAD, wWihF, DPAD, nullptr, 0, bih_f, bhh_f,
                xgF, 4 * H_, ROWS, 4 * H_, DPAD, true, stream);
    launch_gemm(emb, DPAD, wWihB, DPAD, nullptr, 0, bih_b, bhh_b,
                xgB, 4 * H_, ROWS, 4 * H_, DPAD, true, stream);

    // 4) sequential biLSTM: fwd processes t, bwd processes L-1-t
    //    (narrow tile: per-step GEMM is latency-bound, maximize wave parallelism)
    for (int t = 0; t < L_; ++t) {
        launch_gemm(h_bf, H_, wWhhF, H_,
                    xgF + (long)t * 4 * H_, (long)L_ * 4 * H_, nullptr, nullptr,
                    gates, 4 * H_, B_, 4 * H_, H_, false, stream);
        launch_gemm(h_bf + (size_t)B_ * H_, H_, wWhhB, H_,
                    xgB + (long)(L_ - 1 - t) * 4 * H_, (long)L_ * 4 * H_, nullptr, nullptr,
                    gates + (size_t)B_ * 4 * H_, 4 * H_, B_, 4 * H_, H_, false, stream);
        lstm_update<<<(2 * B_ * H_ + 255) / 256, 256, 0, stream>>>(gates, c_ws, h_bf, hs_bf, t);
    }

    // 5) decoder precompute: step-invariant projections of hs (N=48 -> narrow)
    launch_gemm(hs_bf, HS_, wWe1, HS_, nullptr, 0, nullptr, nullptr,
                evbase, NPAD, ROWS, NPAD, HS_, false, stream);
    launch_gemm(hs_bf, HS_, wWa1, HS_, nullptr, 0, nullptr, nullptr,
                bsarg, NPAD, ROWS, NPAD, HS_, false, stream);
    launch_gemm(hs_bf, HS_, wWa2, HS_, nullptr, 0, nullptr, nullptr,
                tproj, NPAD, ROWS, NPAD, HS_, false, stream);

    // 6) sequential greedy decoder: tiny rank-(E-1)/(A-1) corrections + argmax
    for (int i = 0; i < L_; ++i) {
        ev_step<<<1, 64, 0, stream>>>(evbase, We, be, g_trg, evact, out_ev, i);
        arg_step<<<ROWS / 256, 256, 0, stream>>>(bsarg, tproj, Wa, ba, g_arg, evact, out_ar, i);
    }
}
